// BlockTucker_42949672960822
// MI455X (gfx1250) — matrix operand order repaired and checked by reading, not compiled, not run on hardware
//
#include <hip/hip_runtime.h>
#include <hip/hip_bf16.h>

typedef _Float16 v16h __attribute__((ext_vector_type(16)));
typedef _Float16 v8h  __attribute__((ext_vector_type(8)));
typedef float    v8f  __attribute__((ext_vector_type(8)));

#define B_      8192
#define DIN     2048
#define MM      1600
#define CC      20
#define SS      80
#define OUTN    3000
#define SP      96              // t-dim padded 80 -> 96 (3 x 32 K-blocks per s)
#define KBIL    (SS * SP)       // 7680
#define NOUTPAD 3008            // 47 x 64 N-tiles for final GEMM

// ---------------------------------------------------------------------------
// CDNA5 async copy: GLOBAL_LOAD_ASYNC_TO_LDS_B128 (ASYNCcnt-tracked).
// Each lane copies 16B from its global address to its LDS offset.
// Flat pointers to __shared__ carry the wave-relative LDS offset in the
// low 32 bits (LDS aperture layout), which is what VDST expects.
// ---------------------------------------------------------------------------
__device__ __forceinline__ void async_load_b128(const _Float16* gptr, void* lptr) {
  unsigned lds_off = (unsigned)(uintptr_t)lptr;
  asm volatile("global_load_async_to_lds_b128 %0, %1, off"
               :: "v"(lds_off), "v"(gptr)
               : "memory");
}

__device__ __forceinline__ void wait_async0() {
  asm volatile("s_wait_asynccnt 0x0" ::: "memory");
}

// ---------------------------------------------------------------------------
// WMMA A-fragment (f16 16x32), per CDNA5 ISA §7.12.2:
// lane L holds row m=L%16; halves at K = k0 + (L/16)*8 + {0..7} and +16.
// ---------------------------------------------------------------------------
__device__ __forceinline__ v16h load_a_frag(const _Float16* __restrict__ A,
                                            int lda, int row, int k0, int lane) {
  const int hi = lane >> 4;
  const _Float16* p = A + (size_t)row * lda + k0 + hi * 8;
  v8h lo = *(const v8h*)p;
  v8h hh = *(const v8h*)(p + 16);
  v16h r;
#pragma unroll
  for (int i = 0; i < 8; ++i) { r[i] = lo[i]; r[i + 8] = hh[i]; }
  return r;
}

// ---------------------------------------------------------------------------
// f32 -> f16 conversion / packing (8-wide)
// ---------------------------------------------------------------------------
__global__ void cvt_f16_v8(const float* __restrict__ src, _Float16* __restrict__ dst, int n8) {
  int i = blockIdx.x * blockDim.x + threadIdx.x;
  if (i >= n8) return;
  const float4* s = (const float4*)src + (size_t)i * 2;
  float4 lo = s[0], hh = s[1];
  v8h d;
  d[0] = (_Float16)lo.x; d[1] = (_Float16)lo.y; d[2] = (_Float16)lo.z; d[3] = (_Float16)lo.w;
  d[4] = (_Float16)hh.x; d[5] = (_Float16)hh.y; d[6] = (_Float16)hh.z; d[7] = (_Float16)hh.w;
  *((v8h*)dst + i) = d;
}

// Wout [3000,1600] f32 -> [3008,1600] f16, zero-padded rows (8-wide)
__global__ void pack_wout(const float* __restrict__ W, _Float16* __restrict__ dst) {
  int i = blockIdx.x * blockDim.x + threadIdx.x;       // 8-elem chunk id
  const int total8 = NOUTPAD * MM / 8;
  if (i >= total8) return;
  size_t base = (size_t)i * 8;
  int r = (int)(base / MM);
  v8h d;
  if (r < OUTN) {
#pragma unroll
    for (int j = 0; j < 8; ++j) d[j] = (_Float16)W[base + j];
  } else {
#pragma unroll
    for (int j = 0; j < 8; ++j) d[j] = (_Float16)0.f;
  }
  *((v8h*)dst + i) = d;
}

// Wb [c][q][s][80] f32 -> Wbp [c][q][s][96] f16 (t padded with zeros, 8-wide)
__global__ void pack_wbp(const float* __restrict__ Wb, _Float16* __restrict__ dst) {
  int i = blockIdx.x * blockDim.x + threadIdx.x;       // 8-elem chunk id
  const int total8 = CC * SS * SS * SP / 8;
  if (i >= total8) return;
  int kc   = i % (SP / 8);          // 0..11 (8-half chunk within padded t)
  int rest = i / (SP / 8);          // ((c*80+q)*80+s)
  v8h d;
  if (kc < SS / 8) {                // t' = kc*8 .. kc*8+7 < 80 : real data
    const float* s = Wb + (size_t)rest * SS + kc * 8;
#pragma unroll
    for (int j = 0; j < 8; ++j) d[j] = (_Float16)s[j];
  } else {
#pragma unroll
    for (int j = 0; j < 8; ++j) d[j] = (_Float16)0.f;
  }
  *((v8h*)dst + i) = d;
}

// ---------------------------------------------------------------------------
// WMMA GEMM:  out[m,n] = sum_k A[m,k]*W[n,k] + bias[n]
// Block = 8 waves = 128(M) x 64(N) tile. The 64x32 B slab (4KB) is staged
// into double-buffered LDS via global_load_async_to_lds_b128 (one per
// thread per K-step) and shared by all 8 waves; waits amortize to one
// s_wait_asynccnt + barrier per K-step. A fragments are register
// double-buffered one K-step ahead.
// ---------------------------------------------------------------------------
__global__ __launch_bounds__(256) void gemm_wmma(
    const _Float16* __restrict__ A, const _Float16* __restrict__ W,
    const float* __restrict__ bias,
    _Float16* __restrict__ outh, float* __restrict__ outf,
    int n64, int K, int nvalid) {
  __shared__ _Float16 btile[2][64][32];   // 2 x 4KB

  const int tid  = threadIdx.x;
  const int lane = tid & 31;
  const int wave = tid >> 5;
  const int mblk = blockIdx.x / n64;
  const int n0   = (blockIdx.x % n64) * 64;
  const int m16  = lane & 15;
  const int hi   = lane >> 4;
  const int arow = mblk * 128 + wave * 16 + m16;

  // staging assignment: thread t covers B row (t>>2), 8-half chunk (t&3)
  const int brow = tid >> 2;          // 0..63
  const int bkc  = (tid & 3) * 8;     // half offset within 32-K slab
  const _Float16* bsrc = W + (size_t)(n0 + brow) * K + bkc;

  // prologue: stage k0=0, preload first A fragment
  async_load_b128(bsrc, &btile[0][brow][bkc]);
  v16h a_cur = load_a_frag(A, K, arow, 0, lane);
  v16h a_nxt = a_cur;
  wait_async0();
  __syncthreads();

  v8f acc[4] = {};
  for (int k0 = 0; k0 < K; k0 += 32) {
    const int cur = (k0 >> 5) & 1;
    if (k0 + 32 < K) {
      async_load_b128(bsrc + k0 + 32, &btile[cur ^ 1][brow][bkc]);
      a_nxt = load_a_frag(A, K, arow, k0 + 32, lane);
    }
#pragma unroll
    for (int j = 0; j < 4; ++j) {
      v16h b = __builtin_shufflevector(*(const v8h*)&btile[cur][j * 16 + m16][hi * 8], *(const v8h*)&btile[cur][j * 16 + m16][hi * 8 + 16], 0, 1, 2, 3, 4, 5, 6, 7, 8, 9, 10, 11, 12, 13, 14, 15);  // 2x ds_load_b128
      acc[j] = __builtin_amdgcn_wmma_f32_16x16x32_f16(
          false, a_cur, false, b, (short)0, acc[j], false, false);
    }
    wait_async0();
    __syncthreads();
    a_cur = a_nxt;
  }

#pragma unroll
  for (int j = 0; j < 4; ++j) {
    const int n = n0 + j * 16 + m16;
    const float bi = (n < nvalid) ? bias[n] : 0.f;
#pragma unroll
    for (int r = 0; r < 8; ++r) {
      const int m = mblk * 128 + wave * 16 + r + hi * 8;
      float v = acc[j][r] + bi;
      if (outh) {
        outh[(size_t)m * (n64 * 64) + n] = (_Float16)v;
      } else if (n < nvalid) {
        outf[(size_t)m * nvalid + n] = v;
      }
    }
  }
}

// ---------------------------------------------------------------------------
// Bilinear + signed-sqrt + per-chunk L2 normalize.
// z[b,q] = (u[b] (x) v[b]) . vec(Wb[c,q]) as a GEMM with
//   A[m, s*96+t] = u[m,s]*v[m,t] built in registers (3 v-fragments scaled
//   per-s by scalar u[m,s]).
// The per-s Wbp slab (80q x 96k = 15KB) is shared by all 4 waves of the
// block: async-staged into double-buffered LDS, one wait+barrier per s.
// ---------------------------------------------------------------------------
__device__ __forceinline__ void stage_wb(const _Float16* __restrict__ wc,
                                         _Float16* buf, int s, int tid) {
#pragma unroll
  for (int i = 0; i < 8; ++i) {
    int cid = tid + i * 128;                   // 16B chunk id, 960 total
    if (cid < (SS * SP / 8)) {
      int q  = cid / (SP / 8);                 // 0..79
      int kc = cid % (SP / 8);                 // 0..11
      async_load_b128(wc + (size_t)q * KBIL + s * SP + kc * 8, buf + cid * 8);
    }
  }
}

__global__ __launch_bounds__(128) void bilinear_wmma(
    const _Float16* __restrict__ h0, const _Float16* __restrict__ h1,
    const _Float16* __restrict__ wbp, const float* __restrict__ bb,
    _Float16* __restrict__ z) {
  __shared__ _Float16 wtile[2][SS][SP];   // 2 x 15KB
  __shared__ float    zt[4][16][SS];      // 20KB

  const int tid  = threadIdx.x;
  const int lane = tid & 31;
  const int wave = tid >> 5;
  const int c    = blockIdx.x % CC;
  const int bblk = blockIdx.x / CC;
  const int b0   = bblk * 64 + wave * 16;
  const int m16  = lane & 15;
  const int hi   = lane >> 4;

  const _Float16* __restrict__ vrow = h1 + (size_t)(b0 + m16) * MM + c * SS;
  const _Float16* __restrict__ urow = h0 + (size_t)(b0 + m16) * MM + c * SS;
  const _Float16* __restrict__ wc   = wbp + (size_t)c * SS * KBIL;

  // three 32-wide v fragments (t = 0..31, 32..63, 64..95; t>=80 zero pad)
  v16h vf[3];
#pragma unroll
  for (int tb = 0; tb < 2; ++tb) {
    v8h c0 = *(const v8h*)(vrow + tb * 32 + hi * 8);
    v8h c1 = *(const v8h*)(vrow + tb * 32 + hi * 8 + 16);
#pragma unroll
    for (int i = 0; i < 8; ++i) { vf[tb][i] = c0[i]; vf[tb][i + 8] = c1[i]; }
  }
  {
    v8h c0 = *(const v8h*)(vrow + 64 + hi * 8);
#pragma unroll
    for (int i = 0; i < 8; ++i) { vf[2][i] = c0[i]; vf[2][i + 8] = (_Float16)0.f; }
  }

  // prologue: stage s=0
  stage_wb(wc, &wtile[0][0][0], 0, tid);
  wait_async0();
  __syncthreads();

  v8f acc[5] = {};
  for (int s = 0; s < SS; ++s) {
    const int cur = s & 1;
    if (s + 1 < SS) stage_wb(wc, &wtile[cur ^ 1][0][0], s + 1, tid);

    const _Float16 us = urow[s];
#pragma unroll
    for (int tb = 0; tb < 3; ++tb) {
      v16h a = vf[tb] * us;              // outer-product A fragment
#pragma unroll
      for (int j = 0; j < 5; ++j) {
        v16h b = __builtin_shufflevector(*(const v8h*)&wtile[cur][j * 16 + m16][tb * 32 + hi * 8], *(const v8h*)&wtile[cur][j * 16 + m16][tb * 32 + hi * 8 + 16], 0, 1, 2, 3, 4, 5, 6, 7, 8, 9, 10, 11, 12, 13, 14, 15);
        acc[j] = __builtin_amdgcn_wmma_f32_16x16x32_f16(
            false, a, false, b, (short)0, acc[j], false, false);
      }
    }
    wait_async0();
    __syncthreads();
  }

  // bias + signed sqrt -> LDS tile [16 rows][80 q]
#pragma unroll
  for (int j = 0; j < 5; ++j) {
    const int q = j * 16 + m16;
    const float bi = bb[c * SS + q];
#pragma unroll
    for (int r = 0; r < 8; ++r) {
      const int m = r + hi * 8;
      float v = acc[j][r] + bi;
      zt[wave][m][q] = (v >= 0.f) ? __builtin_sqrtf(v) : -__builtin_sqrtf(-v);
    }
  }
  __syncthreads();

  // per-row L2 norm over the 80-wide chunk; lane pair (L, L^16) shares row m16
  float sum = 0.f;
  const int qb = hi * 40;
#pragma unroll
  for (int q = 0; q < 40; ++q) {
    float t = zt[wave][m16][qb + q];
    sum += t * t;
  }
  sum += __shfl_xor(sum, 16, 32);
  const float scale = 1.0f / fmaxf(__builtin_sqrtf(sum), 1e-12f);

  _Float16* __restrict__ zrow = z + (size_t)(b0 + m16) * MM + c * SS;
#pragma unroll
  for (int q = 0; q < 40; ++q)
    zrow[qb + q] = (_Float16)(zt[wave][m16][qb + q] * scale);
}

// ---------------------------------------------------------------------------
extern "C" void kernel_launch(void* const* d_in, const int* in_sizes, int n_in,
                              void* d_out, int out_size, void* d_ws, size_t ws_size,
                              hipStream_t stream) {
  const float* x0   = (const float*)d_in[0];
  const float* x1   = (const float*)d_in[1];
  const float* W0   = (const float*)d_in[2];
  const float* b0   = (const float*)d_in[3];
  const float* W1   = (const float*)d_in[4];
  const float* b1   = (const float*)d_in[5];
  const float* Wb   = (const float*)d_in[6];
  const float* bb   = (const float*)d_in[7];
  const float* Wout = (const float*)d_in[8];
  const float* bout = (const float*)d_in[9];
  float* out        = (float*)d_out;

  char* ws = (char*)d_ws;
  constexpr size_t SZ_XH  = (size_t)B_ * DIN * 2;
  constexpr size_t SZ_WH  = (size_t)MM * DIN * 2;
  constexpr size_t SZ_H   = (size_t)B_ * MM * 2;
  constexpr size_t SZ_WBP = (size_t)CC * SS * KBIL * 2;
  constexpr size_t SZ_WOH = (size_t)NOUTPAD * MM * 2;

  size_t off = 0;
  _Float16* x0h = (_Float16*)(ws + off); off += SZ_XH;
  _Float16* x1h = (_Float16*)(ws + off); off += SZ_XH;
  _Float16* w0h = (_Float16*)(ws + off); off += SZ_WH;
  _Float16* w1h = (_Float16*)(ws + off); off += SZ_WH;
  _Float16* h0h = (_Float16*)(ws + off); off += SZ_H;
  _Float16* h1h = (_Float16*)(ws + off); off += SZ_H;
  _Float16* zh  = (_Float16*)(ws + off); off += SZ_H;
  _Float16* wbp = (_Float16*)(ws + off); off += SZ_WBP;
  _Float16* woh = (_Float16*)(ws + off); off += SZ_WOH;
  (void)ws_size; (void)in_sizes; (void)n_in; (void)out_size;

  const int T = 256;
  // 1) precision conversion + repacking (8-wide)
  cvt_f16_v8<<<(B_ * DIN / 8 + T - 1) / T, T, 0, stream>>>(x0, x0h, B_ * DIN / 8);
  cvt_f16_v8<<<(B_ * DIN / 8 + T - 1) / T, T, 0, stream>>>(x1, x1h, B_ * DIN / 8);
  cvt_f16_v8<<<(MM * DIN / 8 + T - 1) / T, T, 0, stream>>>(W0, w0h, MM * DIN / 8);
  cvt_f16_v8<<<(MM * DIN / 8 + T - 1) / T, T, 0, stream>>>(W1, w1h, MM * DIN / 8);
  pack_wout<<<(NOUTPAD * MM / 8 + T - 1) / T, T, 0, stream>>>(Wout, woh);
  pack_wbp<<<(CC * SS * SS * SP / 8 + T - 1) / T, T, 0, stream>>>(Wb, wbp);

  // 2) h0 = x0 @ W0.T + b0 ; h1 = x1 @ W1.T + b1   (f16 out, f32 accum)
  //    blocks = (8192/128) * (1600/64) = 64 * 25 = 1600
  gemm_wmma<<<1600, 256, 0, stream>>>(x0h, w0h, b0, h0h, nullptr, MM / 64, DIN, MM);
  gemm_wmma<<<1600, 256, 0, stream>>>(x1h, w1h, b1, h1h, nullptr, MM / 64, DIN, MM);

  // 3) block-bilinear + signed sqrt + per-chunk L2 normalize -> zh (f16)
  bilinear_wmma<<<(B_ / 64) * CC, 128, 0, stream>>>(h0h, h1h, wbp, bb, zh);

  // 4) out = z @ Wout.T + bout  (f32 out, N padded to 3008, guarded store)
  //    blocks = 64 * 47 = 3008
  gemm_wmma<<<3008, 256, 0, stream>>>(zh, woh, bout, nullptr, out, NOUTPAD / 64, MM, OUTN);
}
